// GMMEqualVarianceDenoiser_82463372084048
// MI455X (gfx1250) — matrix operand order repaired and checked by reading, not compiled, not run on hardware
//
#include <hip/hip_runtime.h>
#include <hip/hip_bf16.h>
#include <math.h>
#include <stdint.h>

// ---------------------------------------------------------------------------
// GMM equal-variance denoiser for MI455X (gfx1250, wave32, WMMA + TDM)
//   logits = (X @ (scale*means)^T - 0.5*||scale*means||^2) / var_t   [B x N]
//   P      = softmax_rows(logits)                                    [B x N]
//   out    = (dw * X + (1-dw) * (P @ (scale*means))) / scale         [B x D]
// GEMMs run on v_wmma_f32_16x16x32_bf16; GEMM2 tiles are DMA'd by the
// Tensor Data Mover (tensor_load_to_lds, 6-arg clang-23 form) into
// double-buffered LDS.
// ---------------------------------------------------------------------------

typedef __bf16 bf16_t;
typedef __attribute__((ext_vector_type(16))) __bf16 v16bf;
typedef __attribute__((ext_vector_type(8)))  __bf16 v8bf;
typedef __attribute__((ext_vector_type(8)))  float  v8f;
typedef __attribute__((ext_vector_type(4)))  unsigned int u32x4;
typedef __attribute__((ext_vector_type(8)))  int i32x8;
typedef __attribute__((ext_vector_type(4)))  int i32x4;

#define BATCH_N   8192
#define CLUST_N   4096
#define DIM_N     3072
#define KB        32
#define LDST      40          /* padded LDS row stride (bf16): 80B = 20 banks */
#define PROBS_STRIDE 8192     /* bf16 row stride of probs aliased onto f32 logits */

// workspace layout (bytes)
#define WS_SCALARS_OFF ((size_t)0)
#define WS_CSQ_OFF     ((size_t)256)
#define WS_SCT_OFF     ((size_t)32768)
#define WS_LOGITS_OFF  (WS_SCT_OFF + (size_t)DIM_N * CLUST_N * 2)

#ifndef __has_builtin
#define __has_builtin(x) 0
#endif
#if __has_builtin(__builtin_amdgcn_tensor_load_to_lds) && \
    __has_builtin(__builtin_amdgcn_s_wait_tensorcnt)
#define USE_TDM 1
#else
#define USE_TDM 0
#endif

// ---------------------------------------------------------------------------
// helpers
// ---------------------------------------------------------------------------
__device__ __forceinline__ v8bf cvt8(const float4 a, const float4 b) {
  v8bf r;
  r[0]=(__bf16)a.x; r[1]=(__bf16)a.y; r[2]=(__bf16)a.z; r[3]=(__bf16)a.w;
  r[4]=(__bf16)b.x; r[5]=(__bf16)b.y; r[6]=(__bf16)b.z; r[7]=(__bf16)b.w;
  return r;
}
__device__ __forceinline__ v8bf cvt8s(const float4 a, const float4 b, float s) {
  v8bf r;
  r[0]=(__bf16)(a.x*s); r[1]=(__bf16)(a.y*s); r[2]=(__bf16)(a.z*s); r[3]=(__bf16)(a.w*s);
  r[4]=(__bf16)(b.x*s); r[5]=(__bf16)(b.y*s); r[6]=(__bf16)(b.z*s); r[7]=(__bf16)(b.w*s);
  return r;
}
// A-matrix 16x32 bf16 fragment (ISA 7.12.2): lane<16 -> K {0..7,16..23}, lane>=16 -> K {8..15,24..31}
__device__ __forceinline__ v16bf loadA(const bf16_t* tile, int rowBase, int lane) {
  const int row = rowBase + (lane & 15);
  const int kb  = (lane & 16) ? 8 : 0;
  v8bf lo = *(const v8bf*)(tile + row * LDST + kb);
  v8bf hi = *(const v8bf*)(tile + row * LDST + kb + 16);
  return __builtin_shufflevector(lo, hi, 0,1,2,3,4,5,6,7,8,9,10,11,12,13,14,15);
}
// B-matrix 32x16 bf16 fragment: lane<16 -> col=lane, K {0..7,16..23}; lane>=16 -> col=lane-16, K {8..15,24..31}
__device__ __forceinline__ v16bf loadB(const bf16_t* tile, int rowBase, int lane) {
  const int row = rowBase + (lane & 15);   // tile row = B column index
  const int kb  = (lane & 16) ? 8 : 0;
  v8bf lo = *(const v8bf*)(tile + row * LDST + kb);
  v8bf hi = *(const v8bf*)(tile + row * LDST + kb + 16);
  return __builtin_shufflevector(lo, hi, 0,1,2,3,4,5,6,7,8,9,10,11,12,13,14,15);
}

#if USE_TDM
// Issue one TDM 2D tile load: tile_dim0=32 bf16 contiguous elements per row,
// tile_dim1=128 rows, global row stride = stride_elems (bf16 units).
// LDS destination gets 16B of padding after every 64B row -> row stride 80B
// (pad_interval code 3 = 16 DWORDs, pad_amount code 3 = 4 DWORDs), matching LDST=40.
__device__ __forceinline__ void tdm_load_tile(uint32_t lds_off, const bf16_t* gptr,
                                              uint32_t stride_elems) {
  const uint64_t ga = (uint64_t)(uintptr_t)gptr;
  const uint32_t BIG = 0x40000000u;   // huge tensor dims: OOB never triggers
  u32x4 g0;
  g0[0] = 1u;                                         // count=1, user descriptor
  g0[1] = lds_off;                                    // lds_addr (bytes)
  g0[2] = (uint32_t)(ga & 0xFFFFFFFFu);               // global_addr[31:0]
  g0[3] = (uint32_t)((ga >> 32) & 0x1FFFFFFu)         // global_addr[56:32]
        | (2u << 30);                                 // type = 2 ("image")
  i32x8 g1;
  g1[0] = (int)((1u << 16)      // data_size = 1 -> 2 bytes
              | (1u << 20)      // pad_enable
              | (3u << 22)      // pad_interval: 16 DWORDs between pads
              | (3u << 25));    // pad_amount:   4 DWORDs of pad
  g1[1] = (int)((BIG & 0xFFFFu) << 16);                               // tensor_dim0[15:0]
  g1[2] = (int)(((BIG >> 16) & 0xFFFFu) | ((BIG & 0xFFFFu) << 16));   // dim0[31:16] | dim1[15:0]
  g1[3] = (int)(((BIG >> 16) & 0xFFFFu) | (32u << 16));               // dim1[31:16] | tile_dim0=32
  g1[4] = (int)128u;                                                  // tile_dim1=128, tile_dim2=0
  g1[5] = (int)stride_elems;                                          // tensor_dim0_stride[31:0]
  g1[6] = 0;                                                          // stride0[47:32] | stride1[15:0]
  g1[7] = 0;                                                          // stride1[47:16]
  const i32x4 z4 = {0, 0, 0, 0};                                      // groups 2/3 unused (2D)
  const i32x8 z8 = {0, 0, 0, 0, 0, 0, 0, 0};
  // clang-23 / therock-10.0 6-arg form: (g0, g1, g2, g3, extra, cpol)
  __builtin_amdgcn_tensor_load_to_lds(g0, g1, z4, z4, z8, 0);
}
#endif

// ---------------------------------------------------------------------------
// scalar prep: scale, 1/var_t, denoiser weight, 1/scale
// ---------------------------------------------------------------------------
__global__ void scalars_kernel(const float* __restrict__ t, const float* __restrict__ sd,
                               float* __restrict__ s) {
  const float scale = __expf(-t[0]);
  const float s2 = scale * scale;
  const float vt = s2 * sd[0] * sd[0] + (1.0f - s2);
  const float dw = s2 * sd[0] * sd[0] / vt;
  s[0] = scale;
  s[1] = 1.0f / vt;
  s[2] = dw;
  s[3] = 1.0f / scale;
}

// c_sq[n] = scale^2 * ||means_n||^2
__global__ void csq_kernel(const float* __restrict__ means, const float* __restrict__ scalars,
                           float* __restrict__ c_sq) {
  __shared__ float red[256];
  const int n = blockIdx.x;
  float s = 0.f;
  for (int k = threadIdx.x; k < DIM_N; k += 256) {
    const float v = means[(size_t)n * DIM_N + k];
    s += v * v;
  }
  red[threadIdx.x] = s;
  __syncthreads();
  for (int off = 128; off > 0; off >>= 1) {
    if (threadIdx.x < off) red[threadIdx.x] += red[threadIdx.x + off];
    __syncthreads();
  }
  if (threadIdx.x == 0) {
    const float sc = scalars[0];
    c_sq[n] = sc * sc * red[0];
  }
}

// scT[d][n] = bf16(means[n][d] * scale)   (32x32 LDS-tiled transpose)
__global__ void transpose_kernel(const float* __restrict__ means,
                                 const float* __restrict__ scalars,
                                 bf16_t* __restrict__ scT) {
  __shared__ float tile[32][33];
  const float scale = scalars[0];
  const int dT = blockIdx.x * 32;
  const int nT = blockIdx.y * 32;
  const int tx = threadIdx.x & 31;
  const int ty = threadIdx.x >> 5;  // 0..7
#pragma unroll
  for (int i = 0; i < 32; i += 8)
    tile[ty + i][tx] = means[(size_t)(nT + ty + i) * DIM_N + dT + tx];
  __syncthreads();
#pragma unroll
  for (int i = 0; i < 32; i += 8)
    scT[(size_t)(dT + ty + i) * CLUST_N + nT + tx] = (bf16_t)(tile[tx][ty + i] * scale);
}

// ---------------------------------------------------------------------------
// GEMM1: logits[b][n] = (sum_k x[b][k]*sc[n][k] - 0.5*c_sq[n]) / var_t
// f32 -> bf16 conversion in flight; 128x128 tile; 8 waves (2x4); 4x2 WMMAs/wave
// ---------------------------------------------------------------------------
__global__ void __launch_bounds__(256)
gemm1_kernel(const float* __restrict__ x, const float* __restrict__ means,
             const float* __restrict__ scalars, const float* __restrict__ c_sq,
             float* __restrict__ logits) {
  __shared__ __align__(16) bf16_t ldsA[128 * LDST];
  __shared__ __align__(16) bf16_t ldsB[128 * LDST];

  const int tid  = threadIdx.x;
  const int lane = tid & 31;
  const int wave = tid >> 5;
  const int wm = wave & 1;        // 0..1 -> 64-row halves
  const int wn = wave >> 1;       // 0..3 -> 32-col quarters
  const int mT = blockIdx.y * 128;
  const int nT = blockIdx.x * 128;
  const float scale = scalars[0];

  const int lrow  = tid >> 1;         // 0..127 : tile row this thread loads
  const int lpart = (tid & 1) * 16;   // 0 / 16 : k-half this thread loads

  const float* xrow = x     + (size_t)(mT + lrow) * DIM_N + lpart;
  const float* mrow = means + (size_t)(nT + lrow) * DIM_N + lpart;

  v8f acc[4][2] = {};

  for (int kb = 0; kb < DIM_N; kb += KB) {
    __syncthreads();
    {
      const float4* pa = (const float4*)(xrow + kb);
      const float4 a0 = pa[0], a1 = pa[1], a2 = pa[2], a3 = pa[3];
      const float4* pb = (const float4*)(mrow + kb);
      const float4 b0 = pb[0], b1 = pb[1], b2 = pb[2], b3 = pb[3];
      v8bf* da = (v8bf*)(ldsA + lrow * LDST + lpart);
      da[0] = cvt8(a0, a1);
      da[1] = cvt8(a2, a3);
      v8bf* db = (v8bf*)(ldsB + lrow * LDST + lpart);
      db[0] = cvt8s(b0, b1, scale);
      db[1] = cvt8s(b2, b3, scale);
      if (kb + KB < DIM_N) {
        __builtin_prefetch(xrow + kb + KB, 0, 3);
        __builtin_prefetch(mrow + kb + KB, 0, 3);
      }
    }
    __syncthreads();

    v16bf af[4], bfg[2];
#pragma unroll
    for (int fm = 0; fm < 4; ++fm) af[fm] = loadA(ldsA, wm * 64 + fm * 16, lane);
#pragma unroll
    for (int fn = 0; fn < 2; ++fn) bfg[fn] = loadB(ldsB, wn * 32 + fn * 16, lane);
#pragma unroll
    for (int fm = 0; fm < 4; ++fm)
#pragma unroll
      for (int fn = 0; fn < 2; ++fn)
        acc[fm][fn] = __builtin_amdgcn_wmma_f32_16x16x32_bf16(
            false, af[fm], false, bfg[fn], (short)0, acc[fm][fn], false, false);
  }

  const float inv_vt = scalars[1];
#pragma unroll
  for (int fm = 0; fm < 4; ++fm) {
#pragma unroll
    for (int fn = 0; fn < 2; ++fn) {
      const int col  = nT + wn * 32 + fn * 16 + (lane & 15);
      const int row0 = mT + wm * 64 + fm * 16 + ((lane & 16) ? 8 : 0);
      const float csq = c_sq[col];
#pragma unroll
      for (int r = 0; r < 8; ++r)
        logits[(size_t)(row0 + r) * CLUST_N + col] =
            (acc[fm][fn][r] - 0.5f * csq) * inv_vt;
    }
  }
}

// ---------------------------------------------------------------------------
// row softmax over N=4096; writes bf16 probs in place over the f32 row
// ---------------------------------------------------------------------------
__global__ void softmax_kernel(float* __restrict__ logits) {
  __shared__ float red[256];
  const int b = blockIdx.x;
  const int t = threadIdx.x;
  float* row = logits + (size_t)b * CLUST_N;
  bf16_t* prow = (bf16_t*)logits + (size_t)b * PROBS_STRIDE;

  float v[16];
  float mx = -3.402823466e38f;
#pragma unroll
  for (int i = 0; i < 16; ++i) {
    v[i] = row[i * 256 + t];
    mx = fmaxf(mx, v[i]);
  }
  red[t] = mx;
  __syncthreads();
  for (int off = 128; off > 0; off >>= 1) {
    if (t < off) red[t] = fmaxf(red[t], red[t + off]);
    __syncthreads();
  }
  mx = red[0];
  __syncthreads();

  float sum = 0.f;
#pragma unroll
  for (int i = 0; i < 16; ++i) {
    v[i] = __expf(v[i] - mx);
    sum += v[i];
  }
  red[t] = sum;
  __syncthreads();
  for (int off = 128; off > 0; off >>= 1) {
    if (t < off) red[t] += red[t + off];
    __syncthreads();
  }
  const float inv = 1.0f / red[0];
#pragma unroll
  for (int i = 0; i < 16; ++i)
    prow[i * 256 + t] = (bf16_t)(v[i] * inv);
}

// ---------------------------------------------------------------------------
// GEMM2 epilogue helper shared by both variants
// ---------------------------------------------------------------------------
__device__ __forceinline__ void gemm2_epilogue(const v8f acc[4][2], const float* x,
                                               const float* scalars, float* out,
                                               int mT, int dT, int wm, int wn, int lane) {
  const float dw = scalars[2];
  const float inv_scale = scalars[3];
  const float one_m_dw = 1.0f - dw;
#pragma unroll
  for (int fm = 0; fm < 4; ++fm) {
#pragma unroll
    for (int fn = 0; fn < 2; ++fn) {
      const int col  = dT + wn * 32 + fn * 16 + (lane & 15);
      const int row0 = mT + wm * 64 + fm * 16 + ((lane & 16) ? 8 : 0);
#pragma unroll
      for (int r = 0; r < 8; ++r) {
        const size_t idx = (size_t)(row0 + r) * DIM_N + col;
        out[idx] = (dw * x[idx] + one_m_dw * acc[fm][fn][r]) * inv_scale;
      }
    }
  }
}

#if USE_TDM
// ---------------------------------------------------------------------------
// GEMM2 (TDM): out[b][d] = (dw*x[b][d] + (1-dw)*sum_n P[b][n]*sc[n][d]) / scale
// A/B tiles are bf16 copies -> Tensor Data Mover DMA into double-buffered LDS.
// Wave 0 issues 2 TDM loads for buffer (i+1)%2 while all waves run WMMAs on
// buffer i%2; s_wait_tensorcnt(2) keeps the new pair in flight.
// ---------------------------------------------------------------------------
__global__ void __launch_bounds__(256)
gemm2_tdm_kernel(const bf16_t* __restrict__ probs, const bf16_t* __restrict__ scT,
                 const float* __restrict__ x, const float* __restrict__ scalars,
                 float* __restrict__ out) {
  __shared__ __align__(128) bf16_t ldsA[2][128 * LDST];
  __shared__ __align__(128) bf16_t ldsB[2][128 * LDST];

  const int tid  = threadIdx.x;
  const int lane = tid & 31;
  const int wave = tid >> 5;
  const int wm = wave & 1;
  const int wn = wave >> 1;
  const int mT = blockIdx.y * 128;   // batch tile
  const int dT = blockIdx.x * 128;   // feature tile

  const uint32_t ldsA_off[2] = {(uint32_t)(uintptr_t)&ldsA[0][0],
                                (uint32_t)(uintptr_t)&ldsA[1][0]};
  const uint32_t ldsB_off[2] = {(uint32_t)(uintptr_t)&ldsB[0][0],
                                (uint32_t)(uintptr_t)&ldsB[1][0]};
  const bf16_t* aBase = probs + (size_t)mT * PROBS_STRIDE;  // rows: batch
  const bf16_t* bBase = scT   + (size_t)dT * CLUST_N;       // rows: feature d

  v8f acc[4][2] = {};

  // prologue: DMA k-block 0 into buffer 0 (wave-uniform branch: only wave 0 issues)
  if (wave == 0) {
    tdm_load_tile(ldsA_off[0], aBase, PROBS_STRIDE);
    tdm_load_tile(ldsB_off[0], bBase, CLUST_N);
  }

  int cur = 0;
  for (int kb = 0; kb < CLUST_N; kb += KB) {
    __syncthreads();  // everyone done reading buffer (1-cur) before it is rewritten
    if (wave == 0) {
      if (kb + KB < CLUST_N) {
        tdm_load_tile(ldsA_off[1 - cur], aBase + kb + KB, PROBS_STRIDE);
        tdm_load_tile(ldsB_off[1 - cur], bBase + kb + KB, CLUST_N);
        __builtin_amdgcn_s_wait_tensorcnt(2);  // pair for 'cur' complete; new pair in flight
      } else {
        __builtin_amdgcn_s_wait_tensorcnt(0);
      }
    }
    __syncthreads();  // publish buffer 'cur' to all waves

    v16bf af[4], bfg[2];
#pragma unroll
    for (int fm = 0; fm < 4; ++fm) af[fm] = loadA(&ldsA[cur][0], wm * 64 + fm * 16, lane);
#pragma unroll
    for (int fn = 0; fn < 2; ++fn) bfg[fn] = loadB(&ldsB[cur][0], wn * 32 + fn * 16, lane);
#pragma unroll
    for (int fm = 0; fm < 4; ++fm)
#pragma unroll
      for (int fn = 0; fn < 2; ++fn)
        acc[fm][fn] = __builtin_amdgcn_wmma_f32_16x16x32_bf16(
            false, af[fm], false, bfg[fn], (short)0, acc[fm][fn], false, false);
    cur ^= 1;
  }

  gemm2_epilogue(acc, x, scalars, out, mT, dT, wm, wn, lane);
}
#endif  // USE_TDM

// ---------------------------------------------------------------------------
// GEMM2 (fallback): manual global->VGPR->LDS copies, single-buffered
// ---------------------------------------------------------------------------
__global__ void __launch_bounds__(256)
gemm2_kernel(const bf16_t* __restrict__ probs, const bf16_t* __restrict__ scT,
             const float* __restrict__ x, const float* __restrict__ scalars,
             float* __restrict__ out) {
  __shared__ __align__(16) bf16_t ldsA[128 * LDST];
  __shared__ __align__(16) bf16_t ldsB[128 * LDST];

  const int tid  = threadIdx.x;
  const int lane = tid & 31;
  const int wave = tid >> 5;
  const int wm = wave & 1;
  const int wn = wave >> 1;
  const int mT = blockIdx.y * 128;
  const int dT = blockIdx.x * 128;

  const int lrow  = tid >> 1;
  const int lpart = (tid & 1) * 16;

  const bf16_t* arow = probs + (size_t)(mT + lrow) * PROBS_STRIDE + lpart;
  const bf16_t* brow = scT   + (size_t)(dT + lrow) * CLUST_N      + lpart;

  v8f acc[4][2] = {};

  for (int kb = 0; kb < CLUST_N; kb += KB) {
    __syncthreads();
    {
      v8bf a0 = *(const v8bf*)(arow + kb);
      v8bf a1 = *(const v8bf*)(arow + kb + 8);
      v8bf b0 = *(const v8bf*)(brow + kb);
      v8bf b1 = *(const v8bf*)(brow + kb + 8);
      v8bf* da = (v8bf*)(ldsA + lrow * LDST + lpart);
      da[0] = a0; da[1] = a1;
      v8bf* db = (v8bf*)(ldsB + lrow * LDST + lpart);
      db[0] = b0; db[1] = b1;
      if (kb + KB < CLUST_N) {
        __builtin_prefetch(arow + kb + KB, 0, 3);
        __builtin_prefetch(brow + kb + KB, 0, 3);
      }
    }
    __syncthreads();

    v16bf af[4], bfg[2];
#pragma unroll
    for (int fm = 0; fm < 4; ++fm) af[fm] = loadA(ldsA, wm * 64 + fm * 16, lane);
#pragma unroll
    for (int fn = 0; fn < 2; ++fn) bfg[fn] = loadB(ldsB, wn * 32 + fn * 16, lane);
#pragma unroll
    for (int fm = 0; fm < 4; ++fm)
#pragma unroll
      for (int fn = 0; fn < 2; ++fn)
        acc[fm][fn] = __builtin_amdgcn_wmma_f32_16x16x32_bf16(
            false, af[fm], false, bfg[fn], (short)0, acc[fm][fn], false, false);
  }

  gemm2_epilogue(acc, x, scalars, out, mT, dT, wm, wn, lane);
}

// ---------------------------------------------------------------------------
extern "C" void kernel_launch(void* const* d_in, const int* in_sizes, int n_in,
                              void* d_out, int out_size, void* d_ws, size_t ws_size,
                              hipStream_t stream) {
  (void)in_sizes; (void)n_in; (void)out_size; (void)ws_size;
  const float* x     = (const float*)d_in[0];
  const float* t     = (const float*)d_in[1];
  const float* means = (const float*)d_in[2];
  const float* sd    = (const float*)d_in[3];

  char* ws = (char*)d_ws;
  float*  scalars = (float*)(ws + WS_SCALARS_OFF);
  float*  c_sq    = (float*)(ws + WS_CSQ_OFF);
  bf16_t* scT     = (bf16_t*)(ws + WS_SCT_OFF);
  float*  logits  = (float*)(ws + WS_LOGITS_OFF);
  bf16_t* probs   = (bf16_t*)logits;   // bf16 probs aliased over logits rows
  float*  out     = (float*)d_out;

  scalars_kernel<<<1, 1, 0, stream>>>(t, sd, scalars);
  csq_kernel<<<CLUST_N, 256, 0, stream>>>(means, scalars, c_sq);
  transpose_kernel<<<dim3(DIM_N / 32, CLUST_N / 32), 256, 0, stream>>>(means, scalars, scT);
  gemm1_kernel<<<dim3(CLUST_N / 128, BATCH_N / 128), 256, 0, stream>>>(x, means, scalars, c_sq, logits);
  softmax_kernel<<<BATCH_N, 256, 0, stream>>>(logits);
#if USE_TDM
  gemm2_tdm_kernel<<<dim3(DIM_N / 128, BATCH_N / 128), 256, 0, stream>>>(probs, scT, x, scalars, out);
#else
  gemm2_kernel<<<dim3(DIM_N / 128, BATCH_N / 128), 256, 0, stream>>>(probs, scT, x, scalars, out);
#endif
}
